// GraphSAGE_T_65781719105821
// MI455X (gfx1250) — matrix operand order repaired and checked by reading, not compiled, not run on hardware
//
#include <hip/hip_runtime.h>

typedef __attribute__((ext_vector_type(16))) __bf16 v16bf;
typedef __attribute__((ext_vector_type(8)))  float  v8f;

#define NNODES 50000
#define DIM    128

// ---------- helpers ----------
// Native f32->bf16 convert (RNE): lets the backend emit v_cvt_pk_bf16_f32
// instead of a 3-VALU-op bit-twiddle per element.
__device__ __forceinline__ unsigned short f2bf_u(float f) {
  union { __bf16 b; unsigned short s; } c; c.b = (__bf16)f;
  return c.s;
}

// A-fragment: 16-bit A 16x32 layout. lane L: M=L%16 ; half=L/16 selects K
// pattern {half*8+0..7} in v0..v3 and {16+half*8+0..7} in v4..v7.
__device__ __forceinline__ v16bf load_afrag(const float* __restrict__ rowp,
                                            int lid, float sc) {
  const int half = lid >> 4;
  const float* p = rowp + half * 8;
  float4 x0 = *(const float4*)(p + 0);
  float4 x1 = *(const float4*)(p + 4);
  float4 x2 = *(const float4*)(p + 16);
  float4 x3 = *(const float4*)(p + 20);
  float t[16] = {x0.x,x0.y,x0.z,x0.w, x1.x,x1.y,x1.z,x1.w,
                 x2.x,x2.y,x2.z,x2.w, x3.x,x3.y,x3.z,x3.w};
  v16bf a;
#pragma unroll
  for (int i = 0; i < 16; ++i) a[i] = (__bf16)(t[i] * sc);
  return a;
}

// B-fragment from LDS-staged transposed bf16 weights W_T[n][k].
// lane L: N=n_base+L%16 ; K = ko + {(L/16)*8+0..7, 16+(L/16)*8+0..7} (pairs packed per VGPR).
__device__ __forceinline__ v16bf load_bfrag(const unsigned short* __restrict__ sW,
                                            int stride, int n_base, int ko, int lid) {
  const int nl = lid & 15, kh = lid >> 4;
  const unsigned short* p = sW + (size_t)(n_base + nl) * stride + ko + kh * 8;
  uint4 q0 = *(const uint4*)(p);
  uint4 q1 = *(const uint4*)(p + 16);
  union { unsigned int u[8]; v16bf v; } c;
  c.u[0]=q0.x; c.u[1]=q0.y; c.u[2]=q0.z; c.u[3]=q0.w;
  c.u[4]=q1.x; c.u[5]=q1.y; c.u[6]=q1.z; c.u[7]=q1.w;
  return c.v;
}

__device__ __forceinline__ v8f wmma_bf16(v16bf a, v16bf b, v8f c) {
  return __builtin_amdgcn_wmma_f32_16x16x32_bf16(false, a, false, b,
                                                 (short)0, c, false, false);
}

// ---------- small utility kernels ----------
__global__ __launch_bounds__(256) void zero_f32(float* __restrict__ p, long n) {
  long i = (long)blockIdx.x * 256 + threadIdx.x;
  long stride = (long)gridDim.x * 256;
  for (; i < n; i += stride) p[i] = 0.f;
}

__global__ __launch_bounds__(256) void deg_count(const int* __restrict__ dst,
                                                 float* __restrict__ cnt, int E) {
  int e = blockIdx.x * 256 + threadIdx.x;
  if (e < E) atomicAdd(&cnt[dst[e]], 1.0f);
}

__global__ __launch_bounds__(256) void inv_deg(float* __restrict__ cnt, int n) {
  int i = blockIdx.x * 256 + threadIdx.x;
  if (i < n) cnt[i] = 1.0f / fmaxf(cnt[i], 1.0f);
}

__global__ __launch_bounds__(256) void init_out(float* __restrict__ out,
                                                const float* __restrict__ bc2, int n) {
  int i = blockIdx.x * 256 + threadIdx.x;
  if (i < n) out[i] = bc2[0];
}

// one wave per edge: 32 lanes x float4 gather + 4 f32 atomic adds (L2-resident agg)
__global__ __launch_bounds__(256) void edge_scatter(const float* __restrict__ h,
                                                    const int* __restrict__ src,
                                                    const int* __restrict__ dst,
                                                    float* __restrict__ agg, int E) {
  int t = blockIdx.x * 256 + threadIdx.x;
  int e = t >> 5, lane = t & 31;
  if (e >= E) return;
  int s = src[e], d = dst[e];
  float4 v = *(const float4*)(h + (size_t)s * DIM + lane * 4);
  float* p = agg + (size_t)d * DIM + lane * 4;
  atomicAdd(p + 0, v.x); atomicAdd(p + 1, v.y);
  atomicAdd(p + 2, v.z); atomicAdd(p + 3, v.w);
}

// ---------- fused SAGE layer: hout = relu((agg .* inv) @ Wl + hin @ Wr + b) ----------
#define STRIDE_L 136  // bf16 elems; 272B row => 68-dword lane stride, conflict-free
__global__ __launch_bounds__(256) void sage_gemm(const float* __restrict__ agg,
                                                 const float* __restrict__ inv,
                                                 const float* __restrict__ hin,
                                                 const float* __restrict__ Wl,
                                                 const float* __restrict__ Wr,
                                                 const float* __restrict__ bias,
                                                 float* __restrict__ hout, int nrows) {
  __shared__ __align__(16) unsigned short sWl[DIM * STRIDE_L];
  __shared__ __align__(16) unsigned short sWr[DIM * STRIDE_L];
  const int tid = threadIdx.x;
  for (int i = tid; i < DIM * DIM; i += 256) {   // stage W^T as bf16 in LDS
    int k = i >> 7, n = i & 127;
    sWl[(size_t)n * STRIDE_L + k] = f2bf_u(Wl[i]);
    sWr[(size_t)n * STRIDE_L + k] = f2bf_u(Wr[i]);
  }
  __syncthreads();

  const int w = tid >> 5, lid = tid & 31;
  const int m_base = blockIdx.x * 64 + (w >> 1) * 16;
  const int n_base = (w & 1) * 64;
  const int nl = lid & 15, half = lid >> 4;
  int row  = m_base + nl;
  int rowc = row < nrows ? row : nrows - 1;
  const float sc = inv[rowc];
  const float* arow = agg + (size_t)rowc * DIM;
  const float* hrow = hin + (size_t)rowc * DIM;
  __builtin_prefetch(hrow, 0, 0);

  v8f acc[4] = {};
#pragma unroll
  for (int ko = 0; ko < DIM; ko += 32) {         // mean-aggregated branch @ Wl
    v16bf a = load_afrag(arow + ko, lid, sc);
#pragma unroll
    for (int t = 0; t < 4; ++t)
      acc[t] = wmma_bf16(a, load_bfrag(sWl, STRIDE_L, n_base + t * 16, ko, lid), acc[t]);
  }
#pragma unroll
  for (int ko = 0; ko < DIM; ko += 32) {         // self branch @ Wr
    v16bf a = load_afrag(hrow + ko, lid, 1.0f);
#pragma unroll
    for (int t = 0; t < 4; ++t)
      acc[t] = wmma_bf16(a, load_bfrag(sWr, STRIDE_L, n_base + t * 16, ko, lid), acc[t]);
  }

#pragma unroll
  for (int t = 0; t < 4; ++t) {                  // bias + relu + store
    int n = n_base + t * 16 + nl;
    float bn = bias[n];
#pragma unroll
    for (int r = 0; r < 8; ++r) {
      int m = m_base + r + half * 8;
      if (m < nrows) {
        float v = acc[t][r] + bn;
        hout[(size_t)m * DIM + n] = v > 0.f ? v : 0.f;
      }
    }
  }
}

// ---------- edge classifier: out[e] += relu([h[bs],h[bd],feat] @ Wc1 + bc1) . Wc2 ----------
#define STRIDE_C 296  // bf16 elems; 592B row => 148-dword lane stride, conflict-free
__global__ __launch_bounds__(256) void edge_mlp(const float* __restrict__ h,
                                                const int* __restrict__ bs,
                                                const int* __restrict__ bd,
                                                const float* __restrict__ feat,
                                                const float* __restrict__ Wc1,
                                                const float* __restrict__ bc1,
                                                const float* __restrict__ Wc2,
                                                float* __restrict__ out) {
  __shared__ __align__(16) unsigned short sW[DIM * STRIDE_C];
  const int tid = threadIdx.x;
  for (int i = tid; i < 288 * DIM; i += 256) {   // stage Wc1^T as bf16 in LDS
    int k = i >> 7, n = i & 127;
    sW[(size_t)n * STRIDE_C + k] = f2bf_u(Wc1[i]);
  }
  __syncthreads();

  const int w = tid >> 5, lid = tid & 31;
  const int m_base = blockIdx.x * 64 + (w >> 1) * 16;
  const int n_base = (w & 1) * 64;
  const int nl = lid & 15, half = lid >> 4;
  const int row = m_base + nl;                   // 200000 % 64 == 0: no guard needed
  const int is = bs[row], id = bd[row];
  const float* ps = h + (size_t)is * DIM;
  const float* pd = h + (size_t)id * DIM;
  const float* pf = feat + (size_t)row * 32;

  v8f acc[4] = {};
#pragma unroll
  for (int ko = 0; ko < 288; ko += 32) {         // K segments: [0,128)=h[bs], [128,256)=h[bd], [256,288)=feat
    const float* rp = (ko < 128) ? (ps + ko) : ((ko < 256) ? (pd + ko - 128) : pf);
    v16bf a = load_afrag(rp, lid, 1.0f);
#pragma unroll
    for (int t = 0; t < 4; ++t)
      acc[t] = wmma_bf16(a, load_bfrag(sW, STRIDE_C, n_base + t * 16, ko, lid), acc[t]);
  }

  // epilogue: relu(z) dot Wc2 over this wave's 64 columns
  float p[8] = {};
#pragma unroll
  for (int t = 0; t < 4; ++t) {
    int n = n_base + t * 16 + nl;
    float bn = bc1[n], w2 = Wc2[n];
#pragma unroll
    for (int r = 0; r < 8; ++r) {
      float v = acc[t][r] + bn;
      v = v > 0.f ? v : 0.f;
      p[r] += v * w2;
    }
  }
#pragma unroll
  for (int r = 0; r < 8; ++r) {                  // reduce over 16 N-lanes per half-wave
    float v = p[r];
#pragma unroll
    for (int m = 8; m >= 1; m >>= 1) v += __shfl_xor(v, m, 16);
    if (nl == 0) atomicAdd(&out[m_base + r + half * 8], v);
  }
}

// ---------- host ----------
extern "C" void kernel_launch(void* const* d_in, const int* in_sizes, int n_in,
                              void* d_out, int out_size, void* d_ws, size_t ws_size,
                              hipStream_t stream) {
  const float* x    = (const float*)d_in[0];
  const int*   ei   = (const int*)d_in[1];
  const int*   bei  = (const int*)d_in[2];
  const float* feat = (const float*)d_in[3];
  const float* W1l  = (const float*)d_in[4];
  const float* b1   = (const float*)d_in[5];
  const float* W1r  = (const float*)d_in[6];
  const float* W2l  = (const float*)d_in[7];
  const float* b2   = (const float*)d_in[8];
  const float* W2r  = (const float*)d_in[9];
  const float* Wc1  = (const float*)d_in[10];
  const float* bc1  = (const float*)d_in[11];
  const float* Wc2  = (const float*)d_in[12];
  const float* bc2  = (const float*)d_in[13];

  const int E  = in_sizes[1] / 2;   // 800000
  const int EB = in_sizes[2] / 2;   // 200000
  const int N  = NNODES;

  const int* src  = ei;
  const int* dst  = ei + E;
  const int* bsrc = bei;
  const int* bdst = bei + EB;

  float* ws  = (float*)d_ws;
  float* cnt = ws;                              // 50000 (becomes inv-degree)
  float* agg = ws + 65536;                      // 50000x128
  float* h1  = agg + (size_t)N * DIM;           // 50000x128
  float* h2  = h1 + (size_t)N * DIM;            // 50000x128
  float* out = (float*)d_out;

  const long zn = 65536 + (long)N * DIM;        // cnt region + agg
  zero_f32<<<2048, 256, 0, stream>>>(ws, zn);
  deg_count<<<(E + 255) / 256, 256, 0, stream>>>(dst, cnt, E);
  inv_deg<<<(N + 255) / 256, 256, 0, stream>>>(cnt, N);

  const int scat_blocks = (E * 32 + 255) / 256;
  const int gemm_blocks = (N + 63) / 64;

  // layer 1
  edge_scatter<<<scat_blocks, 256, 0, stream>>>(x, src, dst, agg, E);
  sage_gemm<<<gemm_blocks, 256, 0, stream>>>(agg, cnt, x, W1l, W1r, b1, h1, N);
  // layer 2
  zero_f32<<<2048, 256, 0, stream>>>(agg, (long)N * DIM);
  edge_scatter<<<scat_blocks, 256, 0, stream>>>(h1, src, dst, agg, E);
  sage_gemm<<<gemm_blocks, 256, 0, stream>>>(agg, cnt, h1, W2l, W2r, b2, h2, N);
  // classifier
  init_out<<<(EB + 255) / 256, 256, 0, stream>>>(out, bc2, EB);
  edge_mlp<<<EB / 64, 256, 0, stream>>>(h2, bsrc, bdst, feat, Wc1, bc1, Wc2, out);
}
